// TextureFeatureLoss_38457137168598
// MI455X (gfx1250) — compile-verified, matrix operand order checked by reading, not run on hardware
//
#include <hip/hip_runtime.h>
#include <cstdint>
#include <cstddef>

// ---------------------------------------------------------------------------
// GLCM texture-feature loss on gfx1250 (MI455X).
//
// Histogramming as matrix multiply on the WMMA engine:
//   GLCM[i][j] = sum_k (a_k == i) * (b_k == j)  =  (one-hot A)^T x (one-hot B)
// V_WMMA_I32_16X16X64_IU8 performs 64 conflict-free scatter-adds per
// instruction per 16x16 GLCM quadrant, accumulating i32 in VGPRs.
// One-hot operands are built with a SWAR byte-equality trick, and the build
// is software-pipelined (double-buffered) so the independent VALU of the
// next chunk fills the WMMA_IU8 8-cycle WAR hazard window (no v_nop padding).
// ---------------------------------------------------------------------------

typedef int v8i __attribute__((ext_vector_type(8)));

#define GRAY   32
#define IMG_H  512
#define IMG_W  512
#define PITCH  520      // 512 + 8 pad bytes (0xFF) -> tail pairs never match
#define SROWS  64       // stripe height (a-rows); +1 halo row
#define NOFF   4
#define NCHUNK (IMG_W / 64)
#define NFEAT  (64 * NOFF * 3)   // per-input feature count

__device__ __forceinline__ uint64_t ld8(const unsigned char* p) {
  uint64_t w;
  __builtin_memcpy(&w, p, 8);   // LDS load (ds_load), byte-aligned safe
  return w;
}

// SWAR one-hot: byte t of result = 1 iff byte t of w equals the byte
// replicated in rep. Exact per-byte zero detection (no borrow artifacts).
__device__ __forceinline__ uint32_t onehot4(uint32_t w, uint32_t rep) {
  const uint32_t x = w ^ rep;                            // match -> 0x00 byte
  const uint32_t t = (x & 0x7F7F7F7Fu) + 0x7F7F7F7Fu;    // bit7 set iff low7 != 0
  return (~(t | x) & 0x80808080u) >> 7;                  // 0x01 per match
}

// Build the four one-hot WMMA operands for one 64-pair chunk.
// 8-bit A/B operand layout: VGPR v covers K = (v>>1)*16 + half*8 + (v&1)*4 ..+3,
// so VGPR pairs (2v, 2v+1) cover one contiguous 8-byte run -> single ld8.
__device__ __forceinline__ void build_ops(const unsigned char* pa,
                                          const unsigned char* pb,
                                          int k0, int half,
                                          uint32_t rep0, uint32_t rep1,
                                          v8i& a0, v8i& a1, v8i& b0, v8i& b1) {
#pragma unroll
  for (int vv = 0; vv < 4; ++vv) {
    const int kbase = k0 + vv * 16 + half * 8;
    const uint64_t wa = ld8(pa + kbase);
    const uint64_t wb = ld8(pb + kbase);
    const uint32_t wa_lo = (uint32_t)wa, wa_hi = (uint32_t)(wa >> 32);
    const uint32_t wb_lo = (uint32_t)wb, wb_hi = (uint32_t)(wb >> 32);
    a0[2 * vv]     = (int)onehot4(wa_lo, rep0);   // gray 0..15 rows
    a0[2 * vv + 1] = (int)onehot4(wa_hi, rep0);
    a1[2 * vv]     = (int)onehot4(wa_lo, rep1);   // gray 16..31 rows
    a1[2 * vv + 1] = (int)onehot4(wa_hi, rep1);
    b0[2 * vv]     = (int)onehot4(wb_lo, rep0);
    b0[2 * vv + 1] = (int)onehot4(wb_hi, rep0);
    b1[2 * vv]     = (int)onehot4(wb_lo, rep1);
    b1[2 * vv + 1] = (int)onehot4(wb_hi, rep1);
  }
}

__global__ __launch_bounds__(256)
void glcm_features_kernel(const float* __restrict__ x,
                          const float* __restrict__ y,
                          float* __restrict__ feats) {
  __shared__ unsigned char qt[(SROWS + 1) * PITCH];  // quantized stripe (+halo)
  __shared__ int   hist[NOFF * GRAY * GRAY];         // 4 x 32x32 GLCMs
  __shared__ float facc[NOFF * 3];                   // prop accumulators

  const int tid = threadIdx.x;
  const int img = blockIdx.x & 63;
  const int inp = blockIdx.x >> 6;                   // 0 = x, 1 = y
  const float* __restrict__ g =
      (inp == 0 ? x : y) + (size_t)img * IMG_H * IMG_W;

  for (int i = tid; i < NOFF * GRAY * GRAY; i += 256) hist[i] = 0;
  if (tid < NOFF * 3) facc[tid] = 0.0f;
  __syncthreads();

  const int wave = tid >> 5;
  const int lane = tid & 31;
  const int half = lane >> 4;       // lane group (K-range select)
  const int mn   = lane & 15;       // A row index / B column index
  const int off  = wave & 3;        // GLCM offset owned by this wave
  const int sub  = wave >> 2;       // 2 waves per offset interleave rows

  // Replicated SWAR targets; B targets equal A targets (mn, mn+16).
  const uint32_t rep0 = (uint32_t)mn        * 0x01010101u;
  const uint32_t rep1 = (uint32_t)(mn + 16) * 0x01010101u;

  // offsets: angle 0 -> (0,1), pi/4 -> (1,1), pi/2 -> (1,0), 3pi/4 -> (1,-1)
  const int drt[4] = {0, 1, 1, 1};
  const int dct[4] = {1, 1, 0, -1};
  const int dr = drt[off];
  const int dc = dct[off];
  const int ca = (dc < 0) ? 1 : 0;  // a-run start column
  const int cb = (dc > 0) ? 1 : 0;  // b-run start column

  // Per-wave i32 GLCM quadrant accumulators: c[ti][tj], i = 16*ti+M, j = 16*tj+N
  v8i c00 = {}, c01 = {}, c10 = {}, c11 = {};

  for (int s0 = 0; s0 < IMG_H; s0 += SROWS) {
    // ---- cooperative load + quantize stripe rows [s0, s0+SROWS] -----------
    const int nrows = (IMG_H - s0) < (SROWS + 1) ? (IMG_H - s0) : (SROWS + 1);
    for (int i = tid; i < nrows * IMG_W; i += 256) {
      const int r = i >> 9, c = i & (IMG_W - 1);
      const size_t gidx = (size_t)(s0 + r) * IMG_W + c;
      // prefetch next stripe (global_prefetch_b8)
      if (s0 + SROWS < IMG_H)
        __builtin_prefetch(&g[gidx + (size_t)SROWS * IMG_W], 0, 1);
      int q = (int)(g[gidx] * (float)(GRAY - 1));
      q = q < 0 ? 0 : (q > GRAY - 1 ? GRAY - 1 : q);
      qt[r * PITCH + c] = (unsigned char)q;
    }
    // pad columns [512,520) with 0xFF: a dead pair one-hots to all-zero
    for (int i = tid; i < nrows * (PITCH - IMG_W); i += 256) {
      const int r = i / (PITCH - IMG_W), c = i % (PITCH - IMG_W);
      qt[r * PITCH + IMG_W + c] = 0xFFu;
    }
    __syncthreads();

    // ---- WMMA one-hot histogramming (wave-uniform control flow) -----------
    const int avail = (dr == 0) ? (IMG_H - s0) : (IMG_H - 1 - s0);
    const int rlim  = avail < SROWS ? avail : SROWS;
    for (int lr = sub; lr < rlim; lr += 2) {
      const unsigned char* pa = &qt[lr * PITCH + ca];
      const unsigned char* pb = &qt[(lr + dr) * PITCH + cb];

      // Software-pipelined chunk loop: WMMAs on buffer `cur`, then build
      // buffer `cur^1` — the build's VALU fills the WMMA WAR hazard window.
      v8i A0[2], A1[2], B0[2], B1[2];
      build_ops(pa, pb, 0, half, rep0, rep1, A0[0], A1[0], B0[0], B1[0]);
#pragma unroll 2
      for (int it = 0; it < NCHUNK; ++it) {
        const int cur = it & 1;
        c00 = __builtin_amdgcn_wmma_i32_16x16x64_iu8(false, A0[cur], false, B0[cur], c00, false, false);
        c01 = __builtin_amdgcn_wmma_i32_16x16x64_iu8(false, A0[cur], false, B1[cur], c01, false, false);
        c10 = __builtin_amdgcn_wmma_i32_16x16x64_iu8(false, A1[cur], false, B0[cur], c10, false, false);
        c11 = __builtin_amdgcn_wmma_i32_16x16x64_iu8(false, A1[cur], false, B1[cur], c11, false, false);
        if (it + 1 < NCHUNK)
          build_ops(pa, pb, (it + 1) * 64, half, rep0, rep1,
                    A0[cur ^ 1], A1[cur ^ 1], B0[cur ^ 1], B1[cur ^ 1]);
      }
    }
    __syncthreads();  // protect qt before next stripe overwrite
  }

  // ---- merge wave accumulators into shared GLCM (ds_add_u32) --------------
  // C/D layout: VGPR v, lanes 0-15 -> M=v, lanes 16-31 -> M=v+8; N = lane&15
  {
    int* hbase = &hist[off * GRAY * GRAY];
#pragma unroll
    for (int v = 0; v < 8; ++v) {
      const int i0 = v + 8 * half;
      atomicAdd(&hbase[(i0)      * GRAY + mn     ], c00[v]);
      atomicAdd(&hbase[(i0)      * GRAY + mn + 16], c01[v]);
      atomicAdd(&hbase[(i0 + 16) * GRAY + mn     ], c10[v]);
      atomicAdd(&hbase[(i0 + 16) * GRAY + mn + 16], c11[v]);
    }
  }
  __syncthreads();

  // ---- GLCM properties: contrast, energy, homogeneity ---------------------
  // All pixels are valid, so the normalizer is deterministic: S = 2 * Npairs
  const int npairs[4] = {512 * 511, 511 * 511, 511 * 512, 511 * 511};
  for (int o = 0; o < NOFF; ++o) {
    const float invS = 1.0f / (2.0f * (float)npairs[o]);
    const int* hb = &hist[o * GRAY * GRAY];
    float sc = 0.f, se = 0.f, sh = 0.f;
    for (int b = tid; b < GRAY * GRAY; b += 256) {
      const int i = b >> 5, j = b & 31;
      const float p  = (float)(hb[i * GRAY + j] + hb[j * GRAY + i]) * invS;
      const float d2 = (float)((i - j) * (i - j));
      sc += p * d2;
      se += p * p;
      sh += p / (1.0f + d2);
    }
    atomicAdd(&facc[o * 3 + 0], sc);   // ds_add_f32
    atomicAdd(&facc[o * 3 + 1], se);
    atomicAdd(&facc[o * 3 + 2], sh);
  }
  __syncthreads();

  if (tid < NOFF) {
    float* out = &feats[((size_t)inp * 64 + img) * (NOFF * 3) + tid * 3];
    out[0] = facc[tid * 3 + 0];
    out[1] = sqrtf(facc[tid * 3 + 1]);
    out[2] = facc[tid * 3 + 2];
  }
}

__global__ __launch_bounds__(256)
void glcm_loss_kernel(const float* __restrict__ feats, float* __restrict__ out) {
  __shared__ float acc;
  if (threadIdx.x == 0) acc = 0.0f;
  __syncthreads();
  float s = 0.0f;
  for (int i = threadIdx.x; i < NFEAT; i += 256)
    s += fabsf(feats[i] - feats[NFEAT + i]);
  atomicAdd(&acc, s);
  __syncthreads();
  if (threadIdx.x == 0) out[0] = acc / (float)NFEAT;
}

extern "C" void kernel_launch(void* const* d_in, const int* in_sizes, int n_in,
                              void* d_out, int out_size, void* d_ws, size_t ws_size,
                              hipStream_t stream) {
  const float* x = (const float*)d_in[0];
  const float* y = (const float*)d_in[1];
  float* feats = (float*)d_ws;   // [2][64][4][3] = 1536 floats of scratch

  glcm_features_kernel<<<dim3(128), dim3(256), 0, stream>>>(x, y, feats);
  glcm_loss_kernel<<<dim3(1), dim3(256), 0, stream>>>(feats, (float*)d_out);
}
